// LSA_87840671137897
// MI455X (gfx1250) — matrix operand order repaired and checked by reading, not compiled, not run on hardware
//
#include <hip/hip_runtime.h>
#include <hip/hip_bf16.h>
#include <math.h>

// ---------------- problem constants ----------------
#define BB   2
#define TT   2048
#define CC   1024
#define HH   16
#define HS   64
#define MQ   (BB*TT)      // 4096
#define NQKV (3*CC)       // 3072

typedef __bf16 bf16;
typedef __attribute__((ext_vector_type(16))) __bf16 v16bf;
typedef __attribute__((ext_vector_type(8)))  __bf16 v8bf;
typedef __attribute__((ext_vector_type(8)))  float  v8f;

// -------- CDNA5 async global->LDS copy (ASYNCcnt path), with safe fallback ----
#if defined(__has_builtin)
#if __has_builtin(__builtin_amdgcn_global_load_async_to_lds_b128) && \
    __has_builtin(__builtin_amdgcn_s_wait_asynccnt)
#define HAVE_ASYNC_LDS 1
#endif
#endif
#ifndef HAVE_ASYNC_LDS
#define HAVE_ASYNC_LDS 0
#endif

#if HAVE_ASYNC_LDS
// exact parameter pointee type per hipcc diagnostic:
//   '__attribute__((__vector_size__(4 * sizeof(int)))) int __device__ *'
typedef int v4i_mem __attribute__((vector_size(16)));
typedef __attribute__((address_space(1))) v4i_mem gv4i_t;  // global (__device__)
typedef __attribute__((address_space(3))) v4i_mem lv4i_t;  // LDS (__shared__)
#endif

// copy 16 bytes global -> LDS
__device__ __forceinline__ void cp_b128(void* dst_lds, const void* src_g) {
#if HAVE_ASYNC_LDS
  __builtin_amdgcn_global_load_async_to_lds_b128(
      (gv4i_t*)(unsigned long long)src_g,
      (lv4i_t*)(unsigned)(unsigned long long)dst_lds,
      0, 0);
#else
  *(v8bf*)dst_lds = *(const v8bf*)src_g;
#endif
}

__device__ __forceinline__ void wait_async_lds() {
#if HAVE_ASYNC_LDS
  __builtin_amdgcn_s_wait_asynccnt(0);
#endif
}

__device__ __forceinline__ v8f wmma_bf16(v16bf a, v16bf b, v8f c) {
  // D = A(16x32 bf16) * B(32x16 bf16) + C(16x16 f32)
  return __builtin_amdgcn_wmma_f32_16x16x32_bf16(false, a, false, b, (short)0, c,
                                                 false, false);
}

// A-fragment (16x32, MxK), row-major source with leading dim ld (in elements).
// Lanes 0-15: M=lane, K chunks [0..7],[16..23]; lanes 16-31: M=lane-16, K [8..15],[24..31].
__device__ __forceinline__ v16bf load_frag_a(const bf16* base, int ld, int lane) {
  int r  = lane & 15;
  int c0 = (lane < 16) ? 0 : 8;
  union { v16bf v; v8bf h[2]; } u;
  u.h[0] = *(const v8bf*)(base + r * ld + c0);
  u.h[1] = *(const v8bf*)(base + r * ld + c0 + 16);
  return u.v;
}

// B-fragment (32x16, KxN), stored N-major: row = N, contiguous K, leading dim ld.
// Lanes 0-15: N=lane, K chunks [0..7],[16..23]; lanes 16-31: N=lane-16, K [8..15],[24..31].
__device__ __forceinline__ v16bf load_frag_b(const bf16* base, int ld, int lane) {
  int r  = lane & 15;
  int c0 = (lane < 16) ? 0 : 8;
  union { v16bf v; v8bf h[2]; } u;
  u.h[0] = *(const v8bf*)(base + r * ld + c0);
  u.h[1] = *(const v8bf*)(base + r * ld + c0 + 16);
  return u.v;
}

// ---------------- prep kernels ----------------
__global__ __launch_bounds__(256) void cvt_f32_bf16(const float* __restrict__ in,
                                                    bf16* __restrict__ out, int n) {
  int i = blockIdx.x * 256 + threadIdx.x;
  if (i < n) out[i] = (bf16)in[i];
}

// in: f32 [R][Cc] row-major  ->  out: bf16 [Cc][R] (transposed, N-major)
__global__ __launch_bounds__(256) void transpose_cvt(const float* __restrict__ in,
                                                     bf16* __restrict__ out,
                                                     int R, int Cc) {
  __shared__ float tile[32][33];
  int c = blockIdx.x * 32 + threadIdx.x;
  for (int i = threadIdx.y; i < 32; i += 8) {
    int r = blockIdx.y * 32 + i;
    tile[i][threadIdx.x] = in[(size_t)r * Cc + c];
  }
  __syncthreads();
  int rr = blockIdx.y * 32 + threadIdx.x;
  for (int i = threadIdx.y; i < 32; i += 8) {
    int cc = blockIdx.x * 32 + i;
    out[(size_t)cc * R + rr] = (bf16)tile[threadIdx.x][i];
  }
}

// ---------------- generic bf16 WMMA GEMM (double-buffered, BK=64) ----------------
// C[M][N] = A[M][K] * Bt[N][K]^T + bias[N]
// MODE 0: write f32 out[M][N].
// MODE 1: qkv split-store: q,k bf16 [B,H,T,HS] (q scaled 1/sqrt(HS)), v f32.
template <int MODE>
__global__ __launch_bounds__(256) void gemm_bf16(const bf16* __restrict__ A,
                                                 const bf16* __restrict__ Bt,
                                                 const float* __restrict__ bias,
                                                 int M, int N, int K,
                                                 float* __restrict__ out_f32,
                                                 bf16* __restrict__ q_out,
                                                 bf16* __restrict__ k_out,
                                                 float* __restrict__ v_out) {
  __shared__ __align__(16) bf16 As[2][128 * 64];   // 2 x 16 KB
  __shared__ __align__(16) bf16 Bs[2][128 * 64];   // 2 x 16 KB

  const int tid  = threadIdx.x;
  const int wave = tid >> 5;
  const int lane = tid & 31;
  const int wm   = wave & 3;   // 4 waves along M -> 32 rows each
  const int wn   = wave >> 2;  // 2 waves along N -> 64 cols each
  const int m0   = blockIdx.y * 128;
  const int n0   = blockIdx.x * 128;

  v8f acc[2][4];
#pragma unroll
  for (int i = 0; i < 2; ++i)
#pragma unroll
    for (int j = 0; j < 4; ++j) acc[i][j] = (v8f)(0.0f);

  // stage a 128x64 tile pair into buffer `buf` from K-offset `kt`
  auto stage = [&](int buf, int kt) {
#pragma unroll
    for (int i = tid; i < 1024; i += 256) {
      int row = i >> 3, kc = (i & 7) * 8;
      cp_b128(&As[buf][row * 64 + kc], &A[(size_t)(m0 + row) * K + kt + kc]);
      cp_b128(&Bs[buf][row * 64 + kc], &Bt[(size_t)(n0 + row) * K + kt + kc]);
    }
  };

  const int nk = K >> 6;   // K / 64
  stage(0, 0);
  wait_async_lds();
  __syncthreads();

  for (int ik = 0; ik < nk; ++ik) {
    const int buf = ik & 1;
    if (ik + 1 < nk) stage(buf ^ 1, (ik + 1) << 6);   // overlap with compute

#pragma unroll
    for (int kc = 0; kc < 2; ++kc) {                  // two 32-wide K chunks
      v16bf af[2];
#pragma unroll
      for (int sm = 0; sm < 2; ++sm)
        af[sm] = load_frag_a(&As[buf][(wm * 32 + sm * 16) * 64 + kc * 32], 64, lane);
#pragma unroll
      for (int sn = 0; sn < 4; ++sn) {
        v16bf bfr = load_frag_b(&Bs[buf][(wn * 64 + sn * 16) * 64 + kc * 32], 64, lane);
#pragma unroll
        for (int sm = 0; sm < 2; ++sm) acc[sm][sn] = wmma_bf16(af[sm], bfr, acc[sm][sn]);
      }
    }
    wait_async_lds();
    __syncthreads();
  }

  // epilogue: VGPR r, lanes 0-15 -> M=r, lanes 16-31 -> M=8+r; N = lane&15
  const int colHalf = lane & 15;
  const int rOff    = (lane < 16) ? 0 : 8;
#pragma unroll
  for (int sm = 0; sm < 2; ++sm) {
#pragma unroll
    for (int sn = 0; sn < 4; ++sn) {
      int col = n0 + wn * 64 + sn * 16 + colHalf;
      int rowB = m0 + wm * 32 + sm * 16 + rOff;
      float bc = bias[col];
#pragma unroll
      for (int r = 0; r < 8; ++r) {
        int row = rowB + r;
        float val = acc[sm][sn][r] + bc;
        if (MODE == 0) {
          out_f32[(size_t)row * N + col] = val;
        } else {
          int sec = col >> 10;        // 0:q 1:k 2:v
          int cc  = col & 1023;
          int h   = cc >> 6, d = cc & 63;
          int b   = row >> 11, t = row & 2047;
          size_t idx = (((size_t)(b * HH + h)) * TT + t) * HS + d;
          if (sec == 0)      q_out[idx] = (bf16)(val * 0.125f);  // 1/sqrt(64)
          else if (sec == 1) k_out[idx] = (bf16)val;
          else               v_out[idx] = val;
        }
      }
    }
  }
}

// ---------------- per-head gated-V MLP ----------------
// gated_v = (v + v@W1 + b1) * sigmoid(v@W2 + b2); write transposed [B,H,HS,T] bf16
__global__ __launch_bounds__(256) void gate_v(const float* __restrict__ v,
                                              const float* __restrict__ W1,
                                              const float* __restrict__ b1,
                                              const float* __restrict__ W2,
                                              const float* __restrict__ b2,
                                              bf16* __restrict__ gvt) {
  __shared__ float W1s[64 * 64];
  __shared__ float W2s[64 * 64];
  const int tid = threadIdx.x;
  for (int i = tid; i < 4096; i += 256) { W1s[i] = W1[i]; W2s[i] = W2[i]; }
  __syncthreads();

  const int wave = tid >> 5, lane = tid & 31;
  const int ri = blockIdx.x * 8 + wave;        // row index in [0, B*H*T)
  const int bh = ri >> 11, t = ri & 2047;
  const float* vp = v + (size_t)ri * 64;
  float v0 = vp[2 * lane], v1 = vp[2 * lane + 1];

  const int j0 = lane, j1 = lane + 32;
  float a10 = 0.f, a11 = 0.f, a20 = 0.f, a21 = 0.f;
#pragma unroll
  for (int k = 0; k < 64; ++k) {
    float vk = __shfl((k & 1) ? v1 : v0, k >> 1);
    a10 += vk * W1s[k * 64 + j0];
    a11 += vk * W1s[k * 64 + j1];
    a20 += vk * W2s[k * 64 + j0];
    a21 += vk * W2s[k * 64 + j1];
  }
  float vj0 = __shfl((j0 & 1) ? v1 : v0, j0 >> 1);
  float vj1 = __shfl((j1 & 1) ? v1 : v0, j1 >> 1);
  float g0 = 1.0f / (1.0f + expf(-(a20 + b2[j0])));
  float g1 = 1.0f / (1.0f + expf(-(a21 + b2[j1])));
  float o0 = (vj0 + a10 + b1[j0]) * g0;
  float o1 = (vj1 + a11 + b1[j1]) * g1;
  gvt[((size_t)bh * 64 + j0) * TT + t] = (bf16)o0;
  gvt[((size_t)bh * 64 + j1) * TT + t] = (bf16)o1;
}

// ---------------- causal flash attention (double-buffered K/gV tiles) ----------
// grid: (T/128, B*H); block: 256. Each wave owns 16 query rows.
__global__ __launch_bounds__(256) void attn(const bf16* __restrict__ q,
                                            const bf16* __restrict__ k,
                                            const bf16* __restrict__ gvt,
                                            bf16* __restrict__ y) {
  __shared__ __align__(16) bf16 Qs[128 * 64];      // [qrow][hs]        16 KB
  __shared__ __align__(16) bf16 Ks[2][64 * 64];    // [key][hs]  2 x 8 KB
  __shared__ __align__(16) bf16 Vs[2][64 * 64];    // [hs][key]  2 x 8 KB
  __shared__ __align__(16) bf16 Ps[8 * 16 * 64];   // per-wave P        16 KB

  const int tid  = threadIdx.x;
  const int wave = tid >> 5, lane = tid & 31;
  const int q0   = blockIdx.x * 128;
  const int bh   = blockIdx.y;
  const int colHalf = lane & 15;
  const int rOff = (lane < 16) ? 0 : 8;

  auto stageKV = [&](int buf, int kt0) {
    const bf16* kbase = k + ((size_t)bh * TT + kt0) * HS;
#pragma unroll
    for (int i = tid; i < 512; i += 256) {
      int row = i >> 3, c = (i & 7) * 8;
      cp_b128(&Ks[buf][row * 64 + c], &kbase[(size_t)row * HS + c]);
      cp_b128(&Vs[buf][row * 64 + c], &gvt[((size_t)bh * 64 + row) * TT + kt0 + c]);
    }
  };

  // stage Q tile once + first K/gV tile
  const bf16* qbase = q + ((size_t)bh * TT + q0) * HS;
#pragma unroll
  for (int i = tid; i < 1024; i += 256) {
    int row = i >> 3, c = (i & 7) * 8;
    cp_b128(&Qs[row * 64 + c], &qbase[(size_t)row * HS + c]);
  }
  stageKV(0, 0);
  wait_async_lds();
  __syncthreads();

  float mrow[8], lrow[8];
  v8f Oacc[4];
#pragma unroll
  for (int r = 0; r < 8; ++r) { mrow[r] = -1e30f; lrow[r] = 0.0f; }
#pragma unroll
  for (int s = 0; s < 4; ++s) Oacc[s] = (v8f)(0.0f);

  const int nkb = (q0 + 128) / 64;              // causal key-block count
  const int waveRowMax = q0 + wave * 16 + 15;

  for (int kb = 0; kb < nkb; ++kb) {
    const int kt0 = kb * 64;
    const int buf = kb & 1;
    if (kb + 1 < nkb) stageKV(buf ^ 1, (kb + 1) * 64);  // overlap with compute

    if (kt0 <= waveRowMax) {   // else: fully masked for this wave (no barriers inside)
      // ---- S = Q * K^T (16 x 64), f32 accum ----
      v8f s[4];
#pragma unroll
      for (int sn = 0; sn < 4; ++sn) s[sn] = (v8f)(0.0f);
#pragma unroll
      for (int ks = 0; ks < 2; ++ks) {
        v16bf af = load_frag_a(&Qs[(wave * 16) * 64 + ks * 32], 64, lane);
#pragma unroll
        for (int sn = 0; sn < 4; ++sn) {
          v16bf bfr = load_frag_b(&Ks[buf][(sn * 16) * 64 + ks * 32], 64, lane);
          s[sn] = wmma_bf16(af, bfr, s[sn]);
        }
      }

      // ---- causal mask + online softmax ----
      const int rowBase = q0 + wave * 16 + rOff;
      float mnew[8], alpha[8];
#pragma unroll
      for (int r = 0; r < 8; ++r) {
        int row = rowBase + r;
        float mx = -1e30f;
#pragma unroll
        for (int sn = 0; sn < 4; ++sn) {
          int col = kt0 + sn * 16 + colHalf;
          float sv = s[sn][r];
          if (col > row) sv = -1e30f;
          s[sn][r] = sv;
          mx = fmaxf(mx, sv);
        }
#pragma unroll
        for (int off = 1; off < 16; off <<= 1) mx = fmaxf(mx, __shfl_xor(mx, off));
        mnew[r]  = fmaxf(mrow[r], mx);
        alpha[r] = expf(mrow[r] - mnew[r]);
      }
#pragma unroll
      for (int r = 0; r < 8; ++r) {
        float rs = 0.0f;
#pragma unroll
        for (int sn = 0; sn < 4; ++sn) {
          float p = expf(s[sn][r] - mnew[r]);
          s[sn][r] = p;
          rs += p;
        }
#pragma unroll
        for (int off = 1; off < 16; off <<= 1) rs += __shfl_xor(rs, off);
        lrow[r] = lrow[r] * alpha[r] + rs;
        mrow[r] = mnew[r];
#pragma unroll
        for (int sn = 0; sn < 4; ++sn) Oacc[sn][r] *= alpha[r];
      }

      // ---- P (C-layout) -> LDS -> A-layout fragments ----
      bf16* pw = &Ps[wave * 1024];
#pragma unroll
      for (int sn = 0; sn < 4; ++sn)
#pragma unroll
        for (int r = 0; r < 8; ++r)
          pw[(r + rOff) * 64 + sn * 16 + colHalf] = (bf16)s[sn][r];
      asm volatile("" ::: "memory");   // keep LDS stores before reads (HW is in-order)

      // ---- O += P * gatedV ----
#pragma unroll
      for (int ks = 0; ks < 2; ++ks) {
        v16bf af = load_frag_a(pw + ks * 32, 64, lane);
#pragma unroll
        for (int sn = 0; sn < 4; ++sn) {
          v16bf bfr = load_frag_b(&Vs[buf][(sn * 16) * 64 + ks * 32], 64, lane);
          Oacc[sn] = wmma_bf16(af, bfr, Oacc[sn]);
        }
      }
    }

    wait_async_lds();
    __syncthreads();
  }

  // ---- normalize + store y (bf16, [B,T,C]) ----
  const int b = bh >> 4, h = bh & 15;
  const int rowBase = q0 + wave * 16 + rOff;
#pragma unroll
  for (int r = 0; r < 8; ++r) {
    float inv = 1.0f / lrow[r];
    int row = rowBase + r;
#pragma unroll
    for (int sn = 0; sn < 4; ++sn) {
      int hs = sn * 16 + colHalf;
      y[((size_t)b * TT + row) * CC + h * 64 + hs] = (bf16)(Oacc[sn][r] * inv);
    }
  }
}

// ---------------- host launcher ----------------
extern "C" void kernel_launch(void* const* d_in, const int* in_sizes, int n_in,
                              void* d_out, int out_size, void* d_ws, size_t ws_size,
                              hipStream_t stream) {
  const float* x      = (const float*)d_in[0];
  const float* W_attn = (const float*)d_in[1];
  const float* b_attn = (const float*)d_in[2];
  const float* W_l1   = (const float*)d_in[3];
  const float* b_l1   = (const float*)d_in[4];
  const float* W_l2   = (const float*)d_in[5];
  const float* b_l2   = (const float*)d_in[6];
  const float* W_proj = (const float*)d_in[7];
  const float* b_proj = (const float*)d_in[8];
  float* out = (float*)d_out;

  char* ws = (char*)d_ws;
  bf16*  x_bf   = (bf16*)(ws);                  //  8 MB
  bf16*  WqkvT  = (bf16*)(ws + 8388608);        //  6 MB
  bf16*  WprojT = (bf16*)(ws + 14680064);       //  2 MB
  bf16*  qb     = (bf16*)(ws + 16777216);       //  8 MB
  bf16*  kb     = (bf16*)(ws + 25165824);       //  8 MB
  float* vf     = (float*)(ws + 33554432);      // 16 MB
  bf16*  gvt    = (bf16*)(ws + 50331648);       //  8 MB
  bf16*  yb     = (bf16*)(ws + 58720256);       //  8 MB   (total 64 MB)

  // 1) x -> bf16
  cvt_f32_bf16<<<(BB * TT * CC) / 256, 256, 0, stream>>>(x, x_bf, BB * TT * CC);
  // 2) transposed bf16 weights (N-major)
  transpose_cvt<<<dim3(NQKV / 32, CC / 32), dim3(32, 8), 0, stream>>>(W_attn, WqkvT, CC, NQKV);
  transpose_cvt<<<dim3(CC / 32, CC / 32), dim3(32, 8), 0, stream>>>(W_proj, WprojT, CC, CC);
  // 3) qkv GEMM with split store (q scaled, v in f32)
  gemm_bf16<1><<<dim3(NQKV / 128, MQ / 128), 256, 0, stream>>>(
      x_bf, WqkvT, b_attn, MQ, NQKV, CC, nullptr, qb, kb, vf);
  // 4) gated V  -> transposed bf16 [B,H,HS,T]
  gate_v<<<(BB * HH * TT) / 8, 256, 0, stream>>>(vf, W_l1, b_l1, W_l2, b_l2, gvt);
  // 5) causal flash attention -> y bf16 [B,T,C]
  attn<<<dim3(TT / 128, BB * HH), 256, 0, stream>>>(qb, kb, gvt, yb);
  // 6) output projection -> f32 d_out
  gemm_bf16<0><<<dim3(CC / 128, MQ / 128), 256, 0, stream>>>(
      yb, WprojT, b_proj, MQ, CC, CC, out, nullptr, nullptr, nullptr);
}
